// FFWRelativeCrossAttentionModule_47536698032951
// MI455X (gfx1250) — compile-verified, matrix operand order checked by reading, not run on hardware
//
#include <hip/hip_runtime.h>
#include <hip/hip_bf16.h>

// ---------------------------------------------------------------------------
// Types for CDNA5 WMMA (wave32): v_wmma_f32_16x16x32_bf16
// A: 16x32 bf16, B: 32x16 bf16, C/D: 16x16 f32 (8 VGPRs/lane)
// ---------------------------------------------------------------------------
typedef __attribute__((ext_vector_type(16))) __bf16        v16bf;
typedef __attribute__((ext_vector_type(8)))  float         v8f;
typedef __attribute__((ext_vector_type(4)))  unsigned int  v4u;

union FragBF {
    v16bf v;
    v4u   q[2];
};

__device__ __forceinline__ float fast_sigmoid(float x) {
    return 1.0f / (1.0f + __expf(-x));
}

// ---------------------------------------------------------------------------
// One 16 x (NS*16) C strip per wave, NS known at compile time:
// no guards inside the k-loop, statically indexed accumulators, EXEC stays
// all-ones around every WMMA (the only branches are uniform, outside).
// Per k-step: 2 A-frag b128 loads shared by NS B-frag loads / NS independent
// WMMA chains (load:wmma ratio 2.5 at NS=4).
// ---------------------------------------------------------------------------
template <int NS>
__device__ __forceinline__ void gemm_strip(const __hip_bfloat16* __restrict__ arow,
                                           const __hip_bfloat16* __restrict__ wrow,
                                           const float* __restrict__ bias,
                                           float* __restrict__ Cf,
                                           __hip_bfloat16* __restrict__ Cb,
                                           int N, int K, int epi, int outb,
                                           int m0, int nbase, int lo, int hi) {
    v8f acc[NS];
#pragma unroll
    for (int j = 0; j < NS; ++j) acc[j] = v8f{0.f, 0.f, 0.f, 0.f, 0.f, 0.f, 0.f, 0.f};

    for (int k = 0; k < K; k += 32) {
        FragBF a;
        a.q[0] = *reinterpret_cast<const v4u*>(arow + k + hi * 8);
        a.q[1] = *reinterpret_cast<const v4u*>(arow + k + 16 + hi * 8);
#pragma unroll
        for (int j = 0; j < NS; ++j) {
            const __hip_bfloat16* wj = wrow + (size_t)j * 16 * K;
            FragBF b;
            b.q[0] = *reinterpret_cast<const v4u*>(wj + k + hi * 8);
            b.q[1] = *reinterpret_cast<const v4u*>(wj + k + 16 + hi * 8);
            acc[j] = __builtin_amdgcn_wmma_f32_16x16x32_bf16(
                false, a.v, false, b.v, (short)0, acc[j], false, false);
        }
    }

#pragma unroll
    for (int j = 0; j < NS; ++j) {
        const int n = nbase + j * 16 + lo;
        const float bv = (epi != 0 && bias != nullptr) ? bias[n] : 0.f;
#pragma unroll
        for (int v = 0; v < 8; ++v) {
            const int m = m0 + v + 8 * hi;
            float x = acc[j][v];
            if (epi != 0) x += bv;
            if (epi == 2) x = fmaxf(x, 0.f);
            if (epi == 3) x = (x > 20.f) ? x : log1pf(__expf(x));
            const size_t idx = (size_t)m * N + n;
            if (outb) Cb[idx] = __float2bfloat16(x);
            else      Cf[idx] = x;
        }
    }
}

// ---------------------------------------------------------------------------
// WMMA GEMM:  C[M,N] = A[M,K] @ W[N,K]^T   (A row-major, W row-major N x K)
// epi: 0=none, 1=+bias, 2=relu(x+bias), 3=softplus(x+bias)
// outb: 1 -> write bf16 (Cb), 0 -> write f32 (Cf)
// Block = 128 threads (4 waves); grid = (M/16, ceil(N/256))
// ---------------------------------------------------------------------------
__global__ void gemm_wmma_kernel(const __hip_bfloat16* __restrict__ A,
                                 const __hip_bfloat16* __restrict__ W,
                                 const float* __restrict__ bias,
                                 float* __restrict__ Cf,
                                 __hip_bfloat16* __restrict__ Cb,
                                 int M, int N, int K, int epi, int outb) {
    const int wave = threadIdx.x >> 5;
    const int lane = threadIdx.x & 31;
    const int m0 = blockIdx.x * 16;
    const int nbase = (blockIdx.y * 4 + wave) * 64;
    if (nbase >= N) return;  // uniform per wave

    const int lo = lane & 15;
    const int hi = lane >> 4;
    int rem = (N - nbase) >> 4;
    const int nsub = rem > 4 ? 4 : rem;   // uniform per wave

    const __hip_bfloat16* arow = A + (size_t)(m0 + lo) * K;
    const __hip_bfloat16* wrow = W + (size_t)(nbase + lo) * K;

    // uniform dispatch -> each instantiation runs with EXEC all-ones
    if (nsub == 4)      gemm_strip<4>(arow, wrow, bias, Cf, Cb, N, K, epi, outb, m0, nbase, lo, hi);
    else if (nsub == 3) gemm_strip<3>(arow, wrow, bias, Cf, Cb, N, K, epi, outb, m0, nbase, lo, hi);
    else if (nsub == 2) gemm_strip<2>(arow, wrow, bias, Cf, Cb, N, K, epi, outb, m0, nbase, lo, hi);
    else                gemm_strip<1>(arow, wrow, bias, Cf, Cb, N, K, epi, outb, m0, nbase, lo, hi);
}

// ---------------------------------------------------------------------------
// f32 -> bf16 weight conversion
// ---------------------------------------------------------------------------
__global__ void f32_to_bf16_kernel(const float* __restrict__ src,
                                   __hip_bfloat16* __restrict__ dst, long n) {
    long i = (long)blockIdx.x * blockDim.x + threadIdx.x;
    if (i < n) dst[i] = __float2bfloat16(src[i]);
}

// Zero-padded conversion: dst[r*kout+c] = c<kin ? bf16(src[r*srcld+c]) : 0
__global__ void pad_bf16_kernel(const float* __restrict__ src,
                                __hip_bfloat16* __restrict__ dst,
                                long rows, int srcld, int kin, int kout) {
    long i = (long)blockIdx.x * blockDim.x + threadIdx.x;
    long total = rows * (long)kout;
    if (i >= total) return;
    int c = (int)(i % kout);
    long r = i / kout;
    float v = (c < kin) ? src[r * (long)srcld + c] : 0.f;
    dst[i] = __float2bfloat16(v);
}

// ---------------------------------------------------------------------------
// AdaLN modulation: m[b][j] = bias[j] + sum_k silu(t[b][k]) * Wm[j][k]
// ---------------------------------------------------------------------------
__global__ void adaln_mod_kernel(const float* __restrict__ t,
                                 const float* __restrict__ Wm,
                                 const float* __restrict__ bias,
                                 float* __restrict__ m, int B, int D, int twoD) {
    int idx = blockIdx.x * blockDim.x + threadIdx.x;
    if (idx >= B * twoD) return;
    int b = idx / twoD, j = idx % twoD;
    const float* tb = t + (size_t)b * D;
    const float* wr = Wm + (size_t)j * D;
    float acc = bias[j];
    for (int k = 0; k < D; ++k) {
        float tv = tb[k];
        acc += (tv * fast_sigmoid(tv)) * wr[k];
    }
    m[idx] = acc;
}

// aq = q*(1+scale) + shift ; optional f32 and bf16 outputs
__global__ void adaln_apply_kernel(const float* __restrict__ q,
                                   const float* __restrict__ m,
                                   float* __restrict__ outf,
                                   __hip_bfloat16* __restrict__ outb,
                                   long total, int D, long LD) {
    long idx = (long)blockIdx.x * blockDim.x + threadIdx.x;
    if (idx >= total) return;
    int d = (int)(idx % D);
    int b = (int)(idx / LD);
    float sc = m[b * 2 * D + d];
    float sh = m[b * 2 * D + D + d];
    float v = q[idx] * (1.f + sc) + sh;
    if (outf) outf[idx] = v;
    if (outb) outb[idx] = __float2bfloat16(v);
}

// ---------------------------------------------------------------------------
// Depthwise causal conv1d (K=4) + bias + SiLU; reads xi = xz[:, 0:di]
// ---------------------------------------------------------------------------
__global__ void conv_silu_kernel(const float* __restrict__ xz,
                                 const float* __restrict__ cw,
                                 const float* __restrict__ cb,
                                 float* __restrict__ xcf,
                                 __hip_bfloat16* __restrict__ xcb,
                                 int L, int di, long total) {
    long idx = (long)blockIdx.x * blockDim.x + threadIdx.x;
    if (idx >= total) return;
    int c = (int)(idx % di);
    long rl = idx / di;           // b*L + l
    int l = (int)(rl % L);
    float acc = cb[c];
#pragma unroll
    for (int j = 0; j < 4; ++j) {
        int lj = l - 3 + j;
        if (lj >= 0)
            acc += cw[c * 4 + j] * xz[(size_t)(rl + j - 3) * (2 * di) + c];
    }
    float v = acc * fast_sigmoid(acc);
    xcf[idx] = v;
    xcb[idx] = __float2bfloat16(v);
}

// ---------------------------------------------------------------------------
// Selective scan. Block = di (=512) threads, one per channel; grid = B.
// 16-state recurrence in registers; B_t/C_t staged in LDS in 16-step chunks.
// ---------------------------------------------------------------------------
#define SCAN_CHUNK 16
__global__ void mamba_scan_kernel(const float* __restrict__ delta,
                                  const float* __restrict__ xdbl,
                                  const float* __restrict__ xc,
                                  const float* __restrict__ A_log,
                                  float* __restrict__ ys,
                                  int L, int di) {
    __shared__ float sBC[SCAN_CHUNK * 32];
    const int b = blockIdx.x;
    const int ch = threadIdx.x;
    float A[16], h[16];
#pragma unroll
    for (int s = 0; s < 16; ++s) {
        A[s] = -__expf(A_log[ch * 16 + s]);
        h[s] = 0.f;
    }
    for (int l0 = 0; l0 < L; l0 += SCAN_CHUNK) {
        __syncthreads();
        {   // cooperative load of Bm/Cm for the next 16 steps (512 floats)
            int step = threadIdx.x >> 5;
            int j = threadIdx.x & 31;
            sBC[threadIdx.x] =
                xdbl[(size_t)(b * L + l0 + step) * 48 + 16 + j];
        }
        __syncthreads();
        for (int st = 0; st < SCAN_CHUNK; ++st) {
            size_t row = (size_t)(b * L + l0 + st) * di + ch;
            float d = delta[row];
            float x = xc[row];
            float dx = d * x;
            const float* sB = &sBC[st * 32];
            const float* sC = &sBC[st * 32 + 16];
            float y = 0.f;
#pragma unroll
            for (int s = 0; s < 16; ++s) {
                float dA = __expf(d * A[s]);
                h[s] = dA * h[s] + dx * sB[s];
                y += h[s] * sC[s];
            }
            ys[row] = y;
        }
    }
}

// yp = (ys + xc*D_p) * silu(z),  z = xz[:, di:2di];  output bf16 for W_out GEMM
__global__ void ypost_kernel(const float* __restrict__ ys,
                             const float* __restrict__ xc,
                             const float* __restrict__ xz,
                             const float* __restrict__ Dp,
                             __hip_bfloat16* __restrict__ yp,
                             int di, long total) {
    long idx = (long)blockIdx.x * blockDim.x + threadIdx.x;
    if (idx >= total) return;
    int c = (int)(idx % di);
    long rl = idx / di;
    float z = xz[(size_t)rl * (2 * di) + di + c];
    float v = (ys[idx] + xc[idx] * Dp[c]) * (z * fast_sigmoid(z));
    yp[idx] = __float2bfloat16(v);
}

// ---------------------------------------------------------------------------
// Residual + LayerNorm over D=256. One wave per row (8 elems/lane),
// wave32 shuffle reductions. Optional second output (for d_out slices).
// ---------------------------------------------------------------------------
__global__ void ln_kernel(const float* __restrict__ in1,
                          const float* __restrict__ in2,
                          const float* __restrict__ w,
                          const float* __restrict__ bb,
                          float* __restrict__ out,
                          float* __restrict__ out2, int rows) {
    const int wv = threadIdx.x >> 5;
    const int lane = threadIdx.x & 31;
    const int row = blockIdx.x * 8 + wv;
    if (row >= rows) return;
    const float* p1 = in1 + (size_t)row * 256;
    const float* p2 = in2 ? (in2 + (size_t)row * 256) : nullptr;
    float v[8];
    float s = 0.f;
#pragma unroll
    for (int i = 0; i < 8; ++i) {
        int c = lane + 32 * i;
        v[i] = p1[c] + (p2 ? p2[c] : 0.f);
        s += v[i];
    }
    for (int off = 16; off > 0; off >>= 1) s += __shfl_xor(s, off, 32);
    const float mu = s * (1.f / 256.f);
    float s2 = 0.f;
#pragma unroll
    for (int i = 0; i < 8; ++i) { float d = v[i] - mu; s2 += d * d; }
    for (int off = 16; off > 0; off >>= 1) s2 += __shfl_xor(s2, off, 32);
    const float r = rsqrtf(s2 * (1.f / 256.f) + 1e-5f);
#pragma unroll
    for (int i = 0; i < 8; ++i) {
        int c = lane + 32 * i;
        float o = (v[i] - mu) * r * w[c] + bb[c];
        out[(size_t)row * 256 + c] = o;
        if (out2) out2[(size_t)row * 256 + c] = o;
    }
}

// ---------------------------------------------------------------------------
// Host orchestration
// ---------------------------------------------------------------------------
extern "C" void kernel_launch(void* const* d_in, const int* in_sizes, int n_in,
                              void* d_out, int out_size, void* d_ws, size_t ws_size,
                              hipStream_t stream) {
    constexpr int B = 4, L = 4096, D = 256, di = 512, NL = 4, nst = 16, dtr = 16;
    constexpr int M = B * L;                 // 16384 rows
    constexpr long BLD = (long)M * D;        // 4.19M
    constexpr long BLdi = (long)M * di;      // 8.39M
    constexpr long BL2di = (long)M * 2 * di; // 16.8M
    constexpr int twoD = 2 * D;

    const float* q_in    = (const float*)d_in[0];
    // d_in[1] = value (unused by reference forward)
    const float* diff_ts = (const float*)d_in[2];
    const float* W_in    = (const float*)d_in[3];
    const float* conv_w  = (const float*)d_in[4];
    const float* conv_b  = (const float*)d_in[5];
    const float* W_x     = (const float*)d_in[6];
    const float* W_dt    = (const float*)d_in[7];
    const float* b_dt    = (const float*)d_in[8];
    const float* A_log   = (const float*)d_in[9];
    const float* D_p     = (const float*)d_in[10];
    const float* W_out   = (const float*)d_in[11];
    const float* aaW     = (const float*)d_in[12];
    const float* aab     = (const float*)d_in[13];
    const float* alnw    = (const float*)d_in[14];
    const float* alnb    = (const float*)d_in[15];
    const float* fW1     = (const float*)d_in[16];
    const float* fb1     = (const float*)d_in[17];
    const float* fW2     = (const float*)d_in[18];
    const float* fb2     = (const float*)d_in[19];
    const float* flnw    = (const float*)d_in[20];
    const float* flnb    = (const float*)d_in[21];
    const float* faW     = (const float*)d_in[22];
    const float* fab     = (const float*)d_in[23];
    float* out = (float*)d_out;

    // ---- bump allocator over workspace with lifetime-based reuse ----
    char* ws = (char*)d_ws;
    size_t off = 0;
    auto alloc = [&](size_t bytes) -> char* {
        char* p = ws + off;
        off += (bytes + 255) & ~(size_t)255;
        return p;
    };
    float* query  = (float*)alloc(BLD * 4);        // persistent per layer
    float* m_attn = (float*)alloc(B * twoD * 4);
    float* m_ff   = (float*)alloc(B * twoD * 4);
    float* xz     = (float*)alloc(BL2di * 4);      // (M, 1024)
    float* xc_f   = (float*)alloc(BLdi * 4);       // (M, 512)
    __hip_bfloat16* xcb_yp = (__hip_bfloat16*)alloc(BLdi * 2);  // xc bf16, later yp bf16
    float* xdbl   = (float*)alloc((long)M * 48 * 4);
    __hip_bfloat16* dtb = (__hip_bfloat16*)alloc((long)M * 32 * 2);  // dt, K padded 16->32
    float* big2   = (float*)alloc(BLdi * 4);       // delta; later attn(16MB) then ffo(16MB)
    float* big3   = (float*)alloc(BLdi * 4);       // ys; later xf_f + xf_b + h_b
    __hip_bfloat16* aq_b = (__hip_bfloat16*)alloc(BLD * 2);
    __hip_bfloat16* wb_Win = (__hip_bfloat16*)alloc((size_t)2 * di * D * 2);   // 1024x256
    __hip_bfloat16* wb_Wx  = (__hip_bfloat16*)alloc((size_t)48 * di * 2);      // 48x512
    __hip_bfloat16* wb_Wdt = (__hip_bfloat16*)alloc((size_t)di * 32 * 2);      // 512x32 padded
    __hip_bfloat16* wb_Wout= (__hip_bfloat16*)alloc((size_t)D * di * 2);       // 256x512
    __hip_bfloat16* wb_W1  = (__hip_bfloat16*)alloc((size_t)D * D * 2);
    __hip_bfloat16* wb_W2  = (__hip_bfloat16*)alloc((size_t)D * D * 2);
    // aliases
    float* delta = big2;
    float* attn  = big2;                           // after delta dies
    float* ffo   = big2;                           // after attn dies
    float* ysbuf = big3;
    float* xf_f  = big3;                           // after ys dies
    __hip_bfloat16* xf_b = (__hip_bfloat16*)((char*)big3 + BLD * 4);
    __hip_bfloat16* h_b  = (__hip_bfloat16*)((char*)big3 + BLD * 4 + BLD * 2);
    (void)ws_size; (void)in_sizes; (void)n_in; (void)out_size;

    auto eg = [](long n) { return (unsigned)((n + 255) / 256); };  // elementwise grid
    auto gy = [](int N) { return (unsigned)((N + 255) / 256); };   // GEMM grid.y

    // initial query <- input
    hipMemcpyAsync(query, q_in, BLD * 4, hipMemcpyDeviceToDevice, stream);

    for (int i = 0; i < NL; ++i) {
        const float* W_in_i  = W_in  + (size_t)i * 2 * di * D;
        const float* cw_i    = conv_w + (size_t)i * di * 4;
        const float* cb_i    = conv_b + (size_t)i * di;
        const float* W_x_i   = W_x   + (size_t)i * 48 * di;
        const float* W_dt_i  = W_dt  + (size_t)i * di * dtr;
        const float* b_dt_i  = b_dt  + (size_t)i * di;
        const float* A_log_i = A_log + (size_t)i * di * nst;
        const float* D_p_i   = D_p   + (size_t)i * di;
        const float* W_out_i = W_out + (size_t)i * D * di;
        const float* aaW_i = aaW + (size_t)i * twoD * D;
        const float* aab_i = aab + (size_t)i * twoD;
        const float* faW_i = faW + (size_t)i * twoD * D;
        const float* fab_i = fab + (size_t)i * twoD;
        const float* fW1_i = fW1 + (size_t)i * D * D;
        const float* fb1_i = fb1 + (size_t)i * D;
        const float* fW2_i = fW2 + (size_t)i * D * D;
        const float* fb2_i = fb2 + (size_t)i * D;

        // weight -> bf16 (W_dt zero-padded K 16->32)
        f32_to_bf16_kernel<<<eg((long)2 * di * D), 256, 0, stream>>>(W_in_i, wb_Win, (long)2 * di * D);
        f32_to_bf16_kernel<<<eg((long)48 * di), 256, 0, stream>>>(W_x_i, wb_Wx, (long)48 * di);
        pad_bf16_kernel<<<eg((long)di * 32), 256, 0, stream>>>(W_dt_i, wb_Wdt, di, dtr, dtr, 32);
        f32_to_bf16_kernel<<<eg((long)D * di), 256, 0, stream>>>(W_out_i, wb_Wout, (long)D * di);
        f32_to_bf16_kernel<<<eg((long)D * D), 256, 0, stream>>>(fW1_i, wb_W1, (long)D * D);
        f32_to_bf16_kernel<<<eg((long)D * D), 256, 0, stream>>>(fW2_i, wb_W2, (long)D * D);

        // AdaLN modulation vectors (B x 2D)
        adaln_mod_kernel<<<eg((long)B * twoD), 256, 0, stream>>>(diff_ts, aaW_i, aab_i, m_attn, B, D, twoD);
        adaln_mod_kernel<<<eg((long)B * twoD), 256, 0, stream>>>(diff_ts, faW_i, fab_i, m_ff, B, D, twoD);

        // aq (bf16) = adaln(query)
        adaln_apply_kernel<<<eg(BLD), 256, 0, stream>>>(query, m_attn, nullptr, aq_b, BLD, D, (long)L * D);

        // xz = aq @ W_in^T   (M x 1024, K=256)
        gemm_wmma_kernel<<<dim3(M / 16, gy(2 * di)), 128, 0, stream>>>(
            aq_b, wb_Win, nullptr, xz, nullptr, M, 2 * di, D, 0, 0);

        // depthwise causal conv + SiLU
        conv_silu_kernel<<<eg(BLdi), 256, 0, stream>>>(xz, cw_i, cb_i, xc_f, xcb_yp, L, di, BLdi);

        // x_dbl = xc @ W_x^T   (M x 48, K=512); N=48 -> NS=3 strip in wave 0
        gemm_wmma_kernel<<<dim3(M / 16, gy(48)), 128, 0, stream>>>(
            xcb_yp, wb_Wx, nullptr, xdbl, nullptr, M, 48, di, 0, 0);

        // dt bf16, K padded 16->32 (src row stride 48)
        pad_bf16_kernel<<<eg((long)M * 32), 256, 0, stream>>>(xdbl, dtb, M, 48, dtr, 32);

        // delta = softplus(dt @ W_dt^T + b_dt)   (M x 512, K=32)
        gemm_wmma_kernel<<<dim3(M / 16, gy(di)), 128, 0, stream>>>(
            dtb, wb_Wdt, b_dt_i, delta, nullptr, M, di, 32, 3, 0);

        // selective scan
        mamba_scan_kernel<<<B, di, 0, stream>>>(delta, xdbl, xc_f, A_log_i, ysbuf, L, di);

        // yp = (ys + xc*D_p) * silu(z)  -> bf16 (reuses xc bf16 slot)
        ypost_kernel<<<eg(BLdi), 256, 0, stream>>>(ysbuf, xc_f, xz, D_p_i, xcb_yp, di, BLdi);

        // attn = yp @ W_out^T   (M x 256, K=512)
        gemm_wmma_kernel<<<dim3(M / 16, gy(D)), 128, 0, stream>>>(
            xcb_yp, wb_Wout, nullptr, attn, nullptr, M, D, di, 0, 0);

        // query = LN(query + attn)
        ln_kernel<<<M / 8, 256, 0, stream>>>(query, attn, alnw + (size_t)i * D,
                                             alnb + (size_t)i * D, query, nullptr, M);

        // xf (f32 + bf16) = adaln(query)
        adaln_apply_kernel<<<eg(BLD), 256, 0, stream>>>(query, m_ff, xf_f, xf_b, BLD, D, (long)L * D);

        // h = relu(xf @ W1^T + b1)  -> bf16
        gemm_wmma_kernel<<<dim3(M / 16, gy(D)), 128, 0, stream>>>(
            xf_b, wb_W1, fb1_i, nullptr, h_b, M, D, D, 2, 1);

        // ffo = h @ W2^T + b2  -> f32
        gemm_wmma_kernel<<<dim3(M / 16, gy(D)), 128, 0, stream>>>(
            h_b, wb_W2, fb2_i, ffo, nullptr, M, D, D, 1, 0);

        // query = LN(xf + ffo); also write this layer's output slice
        ln_kernel<<<M / 8, 256, 0, stream>>>(xf_f, ffo, flnw + (size_t)i * D,
                                             flnb + (size_t)i * D, query,
                                             out + (size_t)i * BLD, M);
    }
}
